// TriangleAttention_24412594111353
// MI455X (gfx1250) — compile-verified, matrix operand order checked by reading, not run on hardware
//
#include <hip/hip_runtime.h>

typedef __attribute__((ext_vector_type(16))) _Float16 v16h;
typedef __attribute__((ext_vector_type(8)))  _Float16 v8h;
typedef __attribute__((ext_vector_type(4)))  _Float16 v4h;
typedef __attribute__((ext_vector_type(8)))  float    v8f;
typedef __attribute__((ext_vector_type(4)))  float    v4f;

#define II   384
#define JJ   384
#define CC   128
#define HH   4
#define HD   32
#define NROW (II * JJ)            // 147456 positions
#define KT24 (JJ / 16)            // 24 tiles along J
#define QSCALE 0.17677669529663687f   // 1/sqrt(HD)
#define INFV 1000000000.0f
#define EPSV 1e-5f

// ---------------------------------------------------------------------------
// WMMA helper (v_wmma_f32_16x16x32_f16)
// ---------------------------------------------------------------------------
static __device__ __forceinline__ v8f wmma_f16(v16h a, v16h b, v8f c) {
  return __builtin_amdgcn_wmma_f32_16x16x32_f16(false, a, false, b, (short)0, c,
                                                false, false);
}

// Fragment loader: lane-major dim (M for A, N for B) has stride `ld` (f16
// elems), K (32 wide) contiguous from `base`. ISA layout: lane L: idx=L&15,
// khalf=L>>4; elems 0..7 -> K=khalf*8+e, elems 8..15 -> K=16+khalf*8+e.
static __device__ __forceinline__ v16h load_frag16(const _Float16* __restrict__ base,
                                                   int ld) {
  const int lane = threadIdx.x & 31;
  const int m = lane & 15, kh = lane >> 4;
  const _Float16* p = base + (size_t)m * ld + kh * 8;
  v8h lo = *(const v8h*)p;
  v8h hi = *(const v8h*)(p + 16);
  v16h a;
#pragma unroll
  for (int e = 0; e < 8; ++e) { a[e] = lo[e]; a[e + 8] = hi[e]; }
  return a;
}

// ---------------------------------------------------------------------------
// Kernel 0: weight prep. Transpose + convert the five 128x128 f32 weights to
// f16 WT[n][k] (K contiguous -> b128 B-fragment loads), and build a zero-
// padded 16x128 f16 transpose of wz (128x4) for the WMMA triangle bias.
// ---------------------------------------------------------------------------
__global__ void prep_w_kernel(const float* __restrict__ wq, const float* __restrict__ wk,
                              const float* __restrict__ wv, const float* __restrict__ wg,
                              const float* __restrict__ wo, const float* __restrict__ wz,
                              _Float16* __restrict__ wT, _Float16* __restrict__ wzT) {
  const int t = (int)(blockIdx.x * blockDim.x + threadIdx.x);
  if (t < 5 * CC * CC) {
    const int mat = t >> 14, r = t & 16383, n = r >> 7, kk = r & 127;
    const float* W = (mat == 0) ? wq : (mat == 1) ? wk : (mat == 2) ? wv
                     : (mat == 3) ? wg : wo;
    wT[(size_t)mat * CC * CC + n * CC + kk] = (_Float16)W[kk * CC + n];
  } else if (t < 5 * CC * CC + 16 * CC) {
    const int r = t - 5 * CC * CC, n = r >> 7, kk = r & 127;
    wzT[n * CC + kk] = (_Float16)((n < HH) ? wz[kk * HH + n] : 0.0f);
  }
}

// ---------------------------------------------------------------------------
// Kernel 1: LayerNorm over C=128, f32 -> f16. One wave per (i,j) position.
// ---------------------------------------------------------------------------
__global__ void ln_kernel(const float* __restrict__ x, const float* __restrict__ w,
                          const float* __restrict__ b, _Float16* __restrict__ xln) {
  const int lane = threadIdx.x & 31;
  const int row = (int)((blockIdx.x * blockDim.x + threadIdx.x) >> 5);
  if (row >= NROW) return;
  const float* rp = x + (size_t)row * CC;
  v4f vx = *(const v4f*)(rp + lane * 4);
  float s = vx[0] + vx[1] + vx[2] + vx[3];
  float sq = vx[0]*vx[0] + vx[1]*vx[1] + vx[2]*vx[2] + vx[3]*vx[3];
#pragma unroll
  for (int m = 16; m >= 1; m >>= 1) {
    s  += __shfl_xor(s,  m, 32);
    sq += __shfl_xor(sq, m, 32);
  }
  const float mu  = s * (1.0f / CC);
  const float var = sq * (1.0f / CC) - mu * mu;
  const float rstd = rsqrtf(var + EPSV);
  v4h o;
#pragma unroll
  for (int t = 0; t < 4; ++t) {
    const int c = lane * 4 + t;
    o[t] = (_Float16)((vx[t] - mu) * rstd * w[c] + b[c]);
  }
  *(v4h*)(xln + (size_t)row * CC + lane * 4) = o;
}

// ---------------------------------------------------------------------------
// Kernel 2: fused projection pair. A-fragments for the whole K=128 are held
// in registers (32 VGPRs); the two matrices are processed SEQUENTIALLY with
// a single 8-accumulator set (64 VGPRs) -> no spills. 64 v_wmma per wave.
// pairMode 0: matA=q (x QSCALE, [I][H][J][HD]), matB=k ([I][H][J][HD])
// pairMode 1: matA=v (transposed [I][H][HD][J]), matB=g (sigmoid, row-major)
// ---------------------------------------------------------------------------
__global__ __launch_bounds__(128, 1)
void proj_pair_kernel(const _Float16* __restrict__ xln,
                      const _Float16* __restrict__ WTa,
                      const _Float16* __restrict__ WTb,
                      _Float16* __restrict__ outA,
                      _Float16* __restrict__ outB,
                      const float* __restrict__ biasB, int pairMode) {
  const int lane = threadIdx.x & 31;
  const int rowTile = (int)(blockIdx.x * (blockDim.x >> 5) + (threadIdx.x >> 5));
  if (rowTile >= NROW / 16) return;

  v16h a[4];
#pragma unroll
  for (int ks = 0; ks < 4; ++ks)
    a[ks] = load_frag16(xln + (size_t)rowTile * 16 * CC + ks * 32, CC);

  const int n = lane & 15, half = lane >> 4;

#pragma unroll
  for (int mat = 0; mat < 2; ++mat) {
    const _Float16* WT = mat ? WTb : WTa;
    v8f acc[8];
#pragma unroll
    for (int ct = 0; ct < 8; ++ct) acc[ct] = v8f{};
#pragma unroll
    for (int ks = 0; ks < 4; ++ks)
#pragma unroll
      for (int ct = 0; ct < 8; ++ct) {
        v16h b = load_frag16(WT + (size_t)(ct * 16) * CC + ks * 32, CC);
        acc[ct] = wmma_f16(a[ks], b, acc[ct]);
      }

    // epilogue for this matrix
#pragma unroll
    for (int ct = 0; ct < 8; ++ct) {
      const int col = ct * 16 + n;
      const int h = col >> 5, d = col & 31;
#pragma unroll
      for (int e = 0; e < 8; ++e) {
        const int r = rowTile * 16 + e + 8 * half;
        const int i = r / JJ, j = r - i * JJ;
        float val = acc[ct][e];
        size_t idx;
        if (pairMode == 0) {
          if (mat == 0) val *= QSCALE;                         // q (or k)
          idx = ((size_t)(i * HH + h) * JJ + j) * HD + d;
          (mat ? outB : outA)[idx] = (_Float16)val;
        } else {
          if (mat == 0) {                                      // v -> vT
            idx = ((size_t)(i * HH + h) * HD + d) * JJ + j;
            outA[idx] = (_Float16)val;
          } else {                                             // g sigmoid
            val = 1.0f / (1.0f + __expf(-(val + biasB[col])));
            outB[(size_t)r * CC + col] = (_Float16)val;
          }
        }
      }
    }
  }
}

// ---------------------------------------------------------------------------
// Kernel 3: WMMA triangle bias. tri[h][pos] = xln[pos,:]·wz[:,h] + bz[h],
// computed as a 128x16 GEMM against zero-padded wzT (only cols 0..3 stored).
// ---------------------------------------------------------------------------
__global__ void tri_wmma_kernel(const _Float16* __restrict__ xln,
                                const _Float16* __restrict__ wzT,
                                const float* __restrict__ bz,
                                float* __restrict__ tri) {
  const int lane = threadIdx.x & 31;
  const int rowTile = (int)(blockIdx.x * (blockDim.x >> 5) + (threadIdx.x >> 5));
  if (rowTile >= NROW / 16) return;

  v8f acc = {};
#pragma unroll
  for (int ks = 0; ks < 4; ++ks) {
    v16h a = load_frag16(xln + (size_t)rowTile * 16 * CC + ks * 32, CC);
    v16h b = load_frag16(wzT + ks * 32, CC);
    acc = wmma_f16(a, b, acc);
  }
  const int n = lane & 15, half = lane >> 4;
  if (n < HH) {
    const float bzn = bz[n];
#pragma unroll
    for (int e = 0; e < 8; ++e) {
      const int pos = rowTile * 16 + e + 8 * half;
      tri[(size_t)n * NROW + pos] = acc[e] + bzn;
    }
  }
}

// ---------------------------------------------------------------------------
// Kernel 4: attention, two-pass softmax (low register pressure, no spills).
// One wave per (i, h, 16-row q tile).
// Pass 1: 24 WMMAs -> row max only. Pass 2: 24 WMMAs recompute scores,
// exp() -> unnormalized P in LDS + row sums. PV: 24 WMMAs. Normalization is
// folded into the epilogue: the D-layout row mapping (m = e + 8*half) of the
// output tile matches the score-sum distribution exactly, so 1/sum[e] applies
// with no cross-lane traffic. Fused sigmoid gate on store.
// ---------------------------------------------------------------------------
#define LDK (JJ + 8)   // padded f16 stride to dodge bank conflicts

__global__ __launch_bounds__(128, 1)
void attn_kernel(const _Float16* __restrict__ q,
                 const _Float16* __restrict__ k,
                 const _Float16* __restrict__ vT,
                 const _Float16* __restrict__ g,
                 const float* __restrict__ tri,
                 const float* __restrict__ mask,
                 _Float16* __restrict__ og) {
  __shared__ _Float16 plds[4][16][LDK];   // 50176 bytes

  const int lane  = threadIdx.x & 31;
  const int wslot = (int)(threadIdx.x >> 5);
  const int wv    = (int)(blockIdx.x * 4 + wslot);
  if (wv >= II * HH * KT24) return;

  const int qt = wv % KT24;
  const int h  = (wv / KT24) % HH;
  const int i  = wv / (KT24 * HH);

  const int n = lane & 15, half = lane >> 4;

  const _Float16* qbase = q + ((size_t)(i * HH + h) * JJ + qt * 16) * HD;
  const _Float16* kbase = k + (size_t)(i * HH + h) * JJ * HD;
  v16h afrag = load_frag16(qbase, HD);

  // --- pass 1: row max of (QK^T + bias) ---
  float mx[8];
#pragma unroll
  for (int e = 0; e < 8; ++e) mx[e] = -3.0e38f;
#pragma unroll
  for (int kt = 0; kt < KT24; ++kt) {
    v16h bfrag = load_frag16(kbase + (size_t)(kt * 16) * HD, HD);
    v8f c = {};
    c = wmma_f16(afrag, bfrag, c);
    const int kg = kt * 16 + n;
    const float mb = INFV * (mask[(size_t)i * JJ + kg] - 1.0f);
#pragma unroll
    for (int e = 0; e < 8; ++e) {
      const int qg = qt * 16 + e + 8 * half;
      mx[e] = fmaxf(mx[e], c[e] + mb + tri[((size_t)h * JJ + qg) * JJ + kg]);
    }
  }
#pragma unroll
  for (int msk = 1; msk <= 8; msk <<= 1)
#pragma unroll
    for (int e = 0; e < 8; ++e) mx[e] = fmaxf(mx[e], __shfl_xor(mx[e], msk, 32));

  // --- pass 2: recompute scores, exp -> LDS (unnormalized), accumulate sums
  float sum[8];
#pragma unroll
  for (int e = 0; e < 8; ++e) sum[e] = 0.0f;
#pragma unroll
  for (int kt = 0; kt < KT24; ++kt) {
    v16h bfrag = load_frag16(kbase + (size_t)(kt * 16) * HD, HD);
    v8f c = {};
    c = wmma_f16(afrag, bfrag, c);
    const int kg = kt * 16 + n;
    const float mb = INFV * (mask[(size_t)i * JJ + kg] - 1.0f);
#pragma unroll
    for (int e = 0; e < 8; ++e) {
      const int m = e + 8 * half;
      const int qg = qt * 16 + m;
      const float p = __expf(c[e] + mb + tri[((size_t)h * JJ + qg) * JJ + kg]
                             - mx[e]);
      sum[e] += p;
      plds[wslot][m][kt * 16 + n] = (_Float16)p;
    }
  }
#pragma unroll
  for (int msk = 1; msk <= 8; msk <<= 1)
#pragma unroll
    for (int e = 0; e < 8; ++e) sum[e] += __shfl_xor(sum[e], msk, 32);

  // --- O = P_unnorm @ V  (K=384 -> 12 steps, 2 column tiles of HD) ---
  v8f oo[2] = {v8f{}, v8f{}};
  const _Float16* vb = vT + (size_t)(i * HH + h) * HD * JJ;
#pragma unroll
  for (int ks = 0; ks < 12; ++ks) {
    v16h pa;
    {
      const int m = lane & 15;
      v8h lo = *(const v8h*)&plds[wslot][m][ks * 32 + half * 8];
      v8h hi = *(const v8h*)&plds[wslot][m][ks * 32 + 16 + half * 8];
#pragma unroll
      for (int e = 0; e < 8; ++e) { pa[e] = lo[e]; pa[e + 8] = hi[e]; }
    }
#pragma unroll
    for (int nt = 0; nt < 2; ++nt) {
      v16h bfrag = load_frag16(vb + (size_t)(nt * 16) * JJ + ks * 32, JJ);
      oo[nt] = wmma_f16(pa, bfrag, oo[nt]);
    }
  }

  // --- epilogue: normalize rows, apply sigmoid gate, store f16 ---
  float inv[8];
#pragma unroll
  for (int e = 0; e < 8; ++e) inv[e] = 1.0f / sum[e];
#pragma unroll
  for (int nt = 0; nt < 2; ++nt)
#pragma unroll
    for (int e = 0; e < 8; ++e) {
      const int qg  = qt * 16 + e + 8 * half;
      const int col = h * HD + nt * 16 + n;
      const size_t idx = ((size_t)i * JJ + qg) * CC + col;
      og[idx] = (_Float16)(oo[nt][e] * inv[e] * (float)g[idx]);
    }
}

// ---------------------------------------------------------------------------
// Kernel 5: out = og @ wo + bo (f32 out). A-fragments hoisted, 32 WMMAs/wave.
// ---------------------------------------------------------------------------
__global__ __launch_bounds__(128, 1)
void outgemm_kernel(const _Float16* __restrict__ og,
                    const _Float16* __restrict__ woT,
                    const float* __restrict__ bo,
                    float* __restrict__ out) {
  const int lane = threadIdx.x & 31;
  const int rowTile = (int)(blockIdx.x * (blockDim.x >> 5) + (threadIdx.x >> 5));
  if (rowTile >= NROW / 16) return;

  v16h a[4];
#pragma unroll
  for (int ks = 0; ks < 4; ++ks)
    a[ks] = load_frag16(og + (size_t)rowTile * 16 * CC + ks * 32, CC);

  v8f acc[8];
#pragma unroll
  for (int ct = 0; ct < 8; ++ct) acc[ct] = v8f{};
#pragma unroll
  for (int ks = 0; ks < 4; ++ks)
#pragma unroll
    for (int ct = 0; ct < 8; ++ct) {
      v16h b = load_frag16(woT + (size_t)(ct * 16) * CC + ks * 32, CC);
      acc[ct] = wmma_f16(a[ks], b, acc[ct]);
    }

  const int n = lane & 15, half = lane >> 4;
#pragma unroll
  for (int ct = 0; ct < 8; ++ct) {
    const int col = ct * 16 + n;
    const float bias = bo[col];
#pragma unroll
    for (int e = 0; e < 8; ++e) {
      const int r = rowTile * 16 + e + 8 * half;
      out[(size_t)r * CC + col] = acc[ct][e] + bias;
    }
  }
}

// ---------------------------------------------------------------------------
// Host launch
// ---------------------------------------------------------------------------
extern "C" void kernel_launch(void* const* d_in, const int* in_sizes, int n_in,
                              void* d_out, int out_size, void* d_ws, size_t ws_size,
                              hipStream_t stream) {
  const float* x    = (const float*)d_in[0];
  const float* mask = (const float*)d_in[1];
  const float* ln_w = (const float*)d_in[2];
  const float* ln_b = (const float*)d_in[3];
  const float* wz   = (const float*)d_in[4];
  const float* bz   = (const float*)d_in[5];
  const float* wq   = (const float*)d_in[6];
  const float* wk   = (const float*)d_in[7];
  const float* wv   = (const float*)d_in[8];
  const float* wg   = (const float*)d_in[9];
  const float* bg   = (const float*)d_in[10];
  const float* wo   = (const float*)d_in[11];
  const float* bo   = (const float*)d_in[12];
  float* out = (float*)d_out;

  const size_t SZ16 = (size_t)NROW * CC * sizeof(_Float16);  // 37,748,736 B
  char* ws = (char*)d_ws;
  _Float16* xln = (_Float16*)(ws + 0 * SZ16);
  _Float16* qb  = (_Float16*)(ws + 1 * SZ16);
  _Float16* kb  = (_Float16*)(ws + 2 * SZ16);
  _Float16* vTb = (_Float16*)(ws + 3 * SZ16);
  _Float16* gb  = (_Float16*)(ws + 4 * SZ16);
  _Float16* ogb = (_Float16*)(ws + 5 * SZ16);
  float*    tri = (float*)   (ws + 6 * SZ16);                // 2,359,296 B
  _Float16* wT  = (_Float16*)(ws + 6 * SZ16 + 2359296);      // 5 * 32768 B
  _Float16* wzT = (_Float16*)(ws + 6 * SZ16 + 2359296 + 5 * 32768);  // 4096 B

  // 0. Weight prep (transpose + f32->f16; wz zero-padded to 16 cols)
  const int prepN = 5 * CC * CC + 16 * CC;
  prep_w_kernel<<<(prepN + 255) / 256, 256, 0, stream>>>(wq, wk, wv, wg, wo, wz,
                                                         wT, wzT);

  // 1. LayerNorm (one wave / position; 8 waves per 256-thread block)
  ln_kernel<<<NROW / 8, 256, 0, stream>>>(x, ln_w, ln_b, xln);

  // 2. Fused projection pairs (9216 waves each; 4 waves / 128-thread block)
  const int rtBlocks = (NROW / 16) / 4;   // 2304
  proj_pair_kernel<<<rtBlocks, 128, 0, stream>>>(
      xln, wT + 0 * CC * CC, wT + 1 * CC * CC, qb, kb, nullptr, 0);   // q,k
  proj_pair_kernel<<<rtBlocks, 128, 0, stream>>>(
      xln, wT + 2 * CC * CC, wT + 3 * CC * CC, vTb, gb, bg, 1);       // v,g

  // 3. Triangle bias via WMMA
  tri_wmma_kernel<<<rtBlocks, 128, 0, stream>>>(xln, wzT, bz, tri);

  // 4. Attention (36864 waves; 4 waves / block)
  attn_kernel<<<(II * HH * KT24) / 4, 128, 0, stream>>>(qb, kb, vTb, gb, tri,
                                                        mask, ogb);

  // 5. Output projection (32 WMMAs / wave)
  outgemm_kernel<<<rtBlocks, 128, 0, stream>>>(ogb, wT + 4 * CC * CC, bo, out);
}
